// Model_16930761081034
// MI455X (gfx1250) — matrix operand order repaired and checked by reading, not compiled, not run on hardware
//
#include <hip/hip_runtime.h>
#include <hip/hip_bf16.h>

typedef __attribute__((ext_vector_type(16))) _Float16 v16h;
typedef __attribute__((ext_vector_type(8)))  float    v8f;

#define CH 64
#define LDS_PITCH 68   // 64 + 4 pad: banks (4*m + k) % 64 distinct across the 16 rows

// ---------------------------------------------------------------------------
// Init: x = emb[nodes], zero scatter accumulators and degree counters
// ---------------------------------------------------------------------------
__global__ void k_init(const int* __restrict__ nodes, const float* __restrict__ emb,
                       float* __restrict__ x, float* __restrict__ s_out,
                       float* __restrict__ s_back, float* __restrict__ cnt_out,
                       float* __restrict__ cnt_back, int n) {
    int tid = blockIdx.x * 256 + threadIdx.x;
    if (tid < n * CH) {
        x[tid]      = emb[nodes[tid >> 6] * CH + (tid & 63)];
        s_out[tid]  = 0.f;
        s_back[tid] = 0.f;
    }
    if (tid < n) { cnt_out[tid] = 0.f; cnt_back[tid] = 0.f; }
}

// Degree counts (targets for "out" direction, sources for "back" direction)
__global__ void k_count(const int* __restrict__ src, const int* __restrict__ tgt,
                        float* __restrict__ cnt_out, float* __restrict__ cnt_back, int E) {
    int e = blockIdx.x * 256 + threadIdx.x;
    if (e < E) {
        atomicAdd(&cnt_out[tgt[e]], 1.f);
        atomicAdd(&cnt_back[src[e]], 1.f);
    }
}

// counts -> 1/max(cnt,1) in place
__global__ void k_inv(float* __restrict__ cnt_out, float* __restrict__ cnt_back, int n) {
    int i = blockIdx.x * 256 + threadIdx.x;
    if (i < n) {
        cnt_out[i]  = 1.f / fmaxf(cnt_out[i], 1.f);
        cnt_back[i] = 1.f / fmaxf(cnt_back[i], 1.f);
    }
}

// Fold eval-mode BN into an affine: xn = m*scale + shift
__global__ void k_bnfold(const float* __restrict__ gamma, const float* __restrict__ beta,
                         const float* __restrict__ rmean, const float* __restrict__ rvar,
                         float* __restrict__ scale, float* __restrict__ shift, int P) {
    int i = blockIdx.x * 256 + threadIdx.x;
    if (i < P) {
        float s = gamma[i] * rsqrtf(rvar[i] + 1e-5f);
        scale[i] = s;
        shift[i] = beta[i] - rmean[i] * s;
    }
}

// Pack conv weights W[L,2,64,64] into per-lane WMMA B fragments (f16).
// Fragment element j of lane L of tile (ld, nt, kc) = W[ld, nt*16+(L&15), kc*32+16*(L>>4)+j]
__global__ void k_pack_w(const float* __restrict__ W, _Float16* __restrict__ wp, int total) {
    int tid = blockIdx.x * 256 + threadIdx.x;
    if (tid >= total) return;
    int j    = tid & 15;
    int lane = (tid >> 4) & 31;
    int t    = tid >> 9;         // (ld*4 + nt)*2 + kc
    int kc   = t & 1;
    int nt   = (t >> 1) & 3;
    int ld   = t >> 3;           // layer*2 + dir
    int c    = nt * 16 + (lane & 15);
    int k    = kc * 32 + ((lane >> 4) << 4) + j;
    wp[tid]  = (_Float16)W[(ld * CH + c) * CH + k];
}

// Pack Wh[HID,64] into B fragments, tile index = nt*2 + kc
__global__ void k_pack_wh(const float* __restrict__ Wh, _Float16* __restrict__ wp, int total) {
    int tid = blockIdx.x * 256 + threadIdx.x;
    if (tid >= total) return;
    int j    = tid & 15;
    int lane = (tid >> 4) & 31;
    int t    = tid >> 9;         // nt*2 + kc
    int kc   = t & 1;
    int nt   = t >> 1;
    int c    = nt * 16 + (lane & 15);
    int k    = kc * 32 + ((lane >> 4) << 4) + j;
    wp[tid]  = (_Float16)Wh[c * CH + k];
}

// ---------------------------------------------------------------------------
// Edge scatter: one thread per (edge, channel); fully coalesced 256B/edge.
// s_out[t,c] += x[s,c] ; s_back[s,c] += x[t,c]
// ---------------------------------------------------------------------------
__global__ void k_scatter(const float* __restrict__ x, const int* __restrict__ src,
                          const int* __restrict__ tgt, float* __restrict__ s_out,
                          float* __restrict__ s_back, int E) {
    int tid = blockIdx.x * 256 + threadIdx.x;
    if (tid >= E * CH) return;
    int e = tid >> 6;
    int c = tid & 63;
    int s = src[e], t = tgt[e];
    atomicAdd(&s_out[t * CH + c],  x[s * CH + c]);
    atomicAdd(&s_back[s * CH + c], x[t * CH + c]);
}

// ---------------------------------------------------------------------------
// Fused layer update: mean + BN affine -> f16 -> WMMA (64x64 GEMM) + bias
// -> ReLU -> residual add. Block = 256 threads = 8 waves = 128 node rows.
// Consuming threads also re-zero the scatter accumulators for the next layer.
// ---------------------------------------------------------------------------
__global__ void __launch_bounds__(256)
k_layer(float* __restrict__ x, float* __restrict__ s_out, float* __restrict__ s_back,
        const float* __restrict__ inv_out, const float* __restrict__ inv_back,
        const float* __restrict__ scale, const float* __restrict__ shift,
        const float* __restrict__ bias, const _Float16* __restrict__ wpack,
        int layer, int n) {
    __shared__ float lds[128 * LDS_PITCH];
    const int t    = threadIdx.x;
    const int bb   = blockIdx.x * 128;        // first node row of this block
    const int lane = t & 31;
    const int w    = t >> 5;                  // wave in block
    const int m    = lane & 15;               // A row / D column index
    const int hi   = lane >> 4;
    const int mrow = w * 16 + m;              // local row for A fragment

    float acc[4][8];
#pragma unroll
    for (int nt = 0; nt < 4; ++nt)
#pragma unroll
        for (int r = 0; r < 8; ++r) acc[nt][r] = 0.f;

#pragma unroll
    for (int dir = 0; dir < 2; ++dir) {
        float* sacc = dir ? s_back : s_out;
        const float* inv = dir ? inv_back : inv_out;
        const int poff = (layer * 2 + dir) * CH;

        __syncthreads();   // previous direction's LDS reads complete
        // Coalesced stage of 128 rows, BN applied, accumulator re-zeroed
#pragma unroll
        for (int q = 0; q < 8; ++q) {
            int flat4 = t + 256 * q;          // 0..2047 float4 slots
            int rl = flat4 >> 4;              // local row
            int k4 = (flat4 & 15) << 2;       // channel base
            int rg = bb + rl;
            float4 v = make_float4(0.f, 0.f, 0.f, 0.f);
            if (rg < n) {
                float iv = inv[rg];
                const float4 sv = *(const float4*)(sacc + (size_t)rg * CH + k4);
                *(float4*)(sacc + (size_t)rg * CH + k4) = make_float4(0.f, 0.f, 0.f, 0.f);
                v.x = sv.x * iv * scale[poff + k4 + 0] + shift[poff + k4 + 0];
                v.y = sv.y * iv * scale[poff + k4 + 1] + shift[poff + k4 + 1];
                v.z = sv.z * iv * scale[poff + k4 + 2] + shift[poff + k4 + 2];
                v.w = sv.w * iv * scale[poff + k4 + 3] + shift[poff + k4 + 3];
            }
            *(float4*)&lds[rl * LDS_PITCH + k4] = v;
        }
        __syncthreads();

        // Build A fragments (16x32 f16, two K chunks)
        v16h a0, a1;
#pragma unroll
        for (int j = 0; j < 16; ++j) {
            int k0 = j + (hi << 4);
            a0[j] = (_Float16)lds[mrow * LDS_PITCH + k0];
            a1[j] = (_Float16)lds[mrow * LDS_PITCH + k0 + 32];
        }

        const _Float16* wb_base = wpack + (size_t)(layer * 2 + dir) * 4 * 1024;
#pragma unroll
        for (int nt = 0; nt < 4; ++nt) {
            float bv = bias[poff + nt * 16 + m];
            v8f c;
#pragma unroll
            for (int r = 0; r < 8; ++r) c[r] = bv;
            const _Float16* wb = wb_base + (size_t)nt * 1024;
            v16h b0 = *(const v16h*)(wb + lane * 16);
            v16h b1 = *(const v16h*)(wb + 512 + lane * 16);
            c = __builtin_amdgcn_wmma_f32_16x16x32_f16(false, a0, false, b0, (short)0, c, false, false);
            c = __builtin_amdgcn_wmma_f32_16x16x32_f16(false, a1, false, b1, (short)0, c, false, false);
#pragma unroll
            for (int r = 0; r < 8; ++r) acc[nt][r] += fmaxf(c[r], 0.f);
        }
    }

    // Residual writeback: x[row, nt*16 + n] += relu(out) + relu(back)
#pragma unroll
    for (int r = 0; r < 8; ++r) {
        int row2 = bb + w * 16 + r + (hi << 3);
        if (row2 < n) {
#pragma unroll
            for (int nt = 0; nt < 4; ++nt)
                x[(size_t)row2 * CH + nt * 16 + m] += acc[nt][r];
        }
    }
}

// ---------------------------------------------------------------------------
// Readout: h = relu(x[rules] @ Wh^T + bh); out = h @ Wo^T, fused.
// Block = 256 threads = 8 waves = 128 rules. 128 WMMAs per wave.
// ---------------------------------------------------------------------------
__global__ void __launch_bounds__(256)
k_readout(const float* __restrict__ x, const int* __restrict__ rules,
          const _Float16* __restrict__ whpack, const float* __restrict__ bh,
          const float* __restrict__ Wo, float* __restrict__ out, int R, int hid_nt) {
    __shared__ float lds[128 * LDS_PITCH];
    const int t  = threadIdx.x;
    const int bb = blockIdx.x * 128;
#pragma unroll
    for (int q = 0; q < 8; ++q) {
        int flat4 = t + 256 * q;
        int rl = flat4 >> 4;
        int k4 = (flat4 & 15) << 2;
        int rg = bb + rl;
        float4 v = make_float4(0.f, 0.f, 0.f, 0.f);
        if (rg < R) {
            int ri = rules[rg];
            v = *(const float4*)(x + (size_t)ri * CH + k4);
        }
        *(float4*)&lds[rl * LDS_PITCH + k4] = v;
    }
    __syncthreads();

    const int lane = t & 31;
    const int w    = t >> 5;
    const int m    = lane & 15;
    const int hi   = lane >> 4;
    const int mrow = w * 16 + m;

    v16h a0, a1;
#pragma unroll
    for (int j = 0; j < 16; ++j) {
        int k0 = j + (hi << 4);
        a0[j] = (_Float16)lds[mrow * LDS_PITCH + k0];
        a1[j] = (_Float16)lds[mrow * LDS_PITCH + k0 + 32];
    }

    float p[8];
#pragma unroll
    for (int r = 0; r < 8; ++r) p[r] = 0.f;

    for (int nt = 0; nt < hid_nt; ++nt) {
        float bv = bh[nt * 16 + m];
        float wo = Wo[nt * 16 + m];
        v8f c;
#pragma unroll
        for (int r = 0; r < 8; ++r) c[r] = bv;
        const _Float16* wb = whpack + (size_t)nt * 1024;
        v16h b0 = *(const v16h*)(wb + lane * 16);
        v16h b1 = *(const v16h*)(wb + 512 + lane * 16);
        c = __builtin_amdgcn_wmma_f32_16x16x32_f16(false, a0, false, b0, (short)0, c, false, false);
        c = __builtin_amdgcn_wmma_f32_16x16x32_f16(false, a1, false, b1, (short)0, c, false, false);
#pragma unroll
        for (int r = 0; r < 8; ++r) p[r] += fmaxf(c[r], 0.f) * wo;
    }

    // Sum the 16 N-lanes of each D row; lanes 0-15 -> rows r, lanes 16-31 -> rows r+8
#pragma unroll
    for (int r = 0; r < 8; ++r) {
        float v = p[r];
        v += __shfl_xor(v, 1, 16);
        v += __shfl_xor(v, 2, 16);
        v += __shfl_xor(v, 4, 16);
        v += __shfl_xor(v, 8, 16);
        if (m == 0) {
            int rg = bb + w * 16 + r + (hi << 3);
            if (rg < R) out[rg] = v;
        }
    }
}

// ---------------------------------------------------------------------------
extern "C" void kernel_launch(void* const* d_in, const int* in_sizes, int n_in,
                              void* d_out, int out_size, void* d_ws, size_t ws_size,
                              hipStream_t stream) {
    const int*   nodes   = (const int*)  d_in[0];
    const int*   sources = (const int*)  d_in[1];
    const int*   targets = (const int*)  d_in[2];
    const int*   rules   = (const int*)  d_in[3];
    const float* emb     = (const float*)d_in[4];
    const float* W       = (const float*)d_in[5];
    const float* b       = (const float*)d_in[6];
    const float* gamma   = (const float*)d_in[7];
    const float* beta    = (const float*)d_in[8];
    const float* rmean   = (const float*)d_in[9];
    const float* rvar    = (const float*)d_in[10];
    const float* Wh      = (const float*)d_in[11];
    const float* bh      = (const float*)d_in[12];
    const float* Wo      = (const float*)d_in[13];
    float* out = (float*)d_out;

    const int n = in_sizes[0];
    const int E = in_sizes[1];
    const int R = in_sizes[3];
    const int L = in_sizes[5] / (2 * CH * CH);   // layers
    const int HID = in_sizes[11] / CH;
    const int hid_nt = HID / 16;

    // Workspace carve (256B aligned slices)
    char* p = (char*)d_ws;
    auto carve = [&](size_t bytes) -> void* {
        void* r = (void*)p;
        p += (bytes + 255) & ~(size_t)255;
        return r;
    };
    float* x       = (float*)carve((size_t)n * CH * 4);
    float* s_out   = (float*)carve((size_t)n * CH * 4);
    float* s_back  = (float*)carve((size_t)n * CH * 4);
    float* inv_out = (float*)carve((size_t)n * 4);    // counts, then inverted in place
    float* inv_back= (float*)carve((size_t)n * 4);
    float* scale   = (float*)carve((size_t)L * 2 * CH * 4);
    float* shift   = (float*)carve((size_t)L * 2 * CH * 4);
    _Float16* wpack  = (_Float16*)carve((size_t)L * 2 * 4 * 2 * 512 * 2);
    _Float16* whpack = (_Float16*)carve((size_t)hid_nt * 2 * 512 * 2);

    const int wpack_total  = L * 2 * 4 * 2 * 512;
    const int whpack_total = hid_nt * 2 * 512;
    const int P = L * 2 * CH;

    k_init<<<(n * CH + 255) / 256, 256, 0, stream>>>(nodes, emb, x, s_out, s_back,
                                                     inv_out, inv_back, n);
    k_count<<<(E + 255) / 256, 256, 0, stream>>>(sources, targets, inv_out, inv_back, E);
    k_inv<<<(n + 255) / 256, 256, 0, stream>>>(inv_out, inv_back, n);
    k_bnfold<<<(P + 255) / 256, 256, 0, stream>>>(gamma, beta, rmean, rvar, scale, shift, P);
    k_pack_w<<<(wpack_total + 255) / 256, 256, 0, stream>>>(W, wpack, wpack_total);
    k_pack_wh<<<(whpack_total + 255) / 256, 256, 0, stream>>>(Wh, whpack, whpack_total);

    const int scat_blocks  = (E * CH + 255) / 256;
    const int layer_blocks = (n + 127) / 128;
    for (int l = 0; l < L; ++l) {
        k_scatter<<<scat_blocks, 256, 0, stream>>>(x, sources, targets, s_out, s_back, E);
        k_layer<<<layer_blocks, 256, 0, stream>>>(x, s_out, s_back, inv_out, inv_back,
                                                  scale, shift, b, wpack, l, n);
    }

    k_readout<<<(R + 127) / 128, 256, 0, stream>>>(x, rules, whpack, bh, Wo, out, R, hid_nt);
}
